// MultiheadCosformerAttention2dSuper_75445395521551
// MI455X (gfx1250) — matrix operand order repaired and checked by reading, not compiled, not run on hardware
//
#include <hip/hip_runtime.h>
#include <hip/hip_bf16.h>

typedef __attribute__((ext_vector_type(16))) __bf16 v16bf;
typedef __attribute__((ext_vector_type(8)))  __bf16 v8bf;
typedef __attribute__((ext_vector_type(8)))  float  v8f;

#define HEADS 8
#define HD    32
#define CDIM  256
#define BD    64
#define NTOK  1024
#define NS    256     // Ns = (H/2)*(W/2)
#define WW    32
#define DP    128     // 4*HD (reweighted dim)
#define BH    512     // BD*HEADS
#define PI2   1.57079632679489662f

// ---- WMMA bf16 16x16x32 fragment helpers (ISA 7.12.2 layouts) --------------
// A 16x32: lane<16 holds K kb+{0..7,16..23} with kb=0; lane>=16 kb=8.
// B 32x16: same K order as A: lane<16 holds K {0..7,16..23}; lane>=16 K {8..15,24..31}.
__device__ __forceinline__ v16bf cat8(v8bf lo, v8bf hi) {
  return __builtin_shufflevector(lo, hi, 0, 1, 2, 3, 4, 5, 6, 7,
                                 8, 9, 10, 11, 12, 13, 14, 15);
}
__device__ __forceinline__ v16bf a_frag_bf16(const __bf16* row, int lane, int k0) {
  const int kb = k0 + ((lane < 16) ? 0 : 8);
  v8bf lo = *(const v8bf*)(row + kb);
  v8bf hi = *(const v8bf*)(row + kb + 16);
  return cat8(lo, hi);
}
__device__ __forceinline__ v16bf b_frag_bf16(const __bf16* row, int lane, int k0) {
  const int kb = k0 + ((lane < 16) ? 0 : 8);
  v8bf lo = *(const v8bf*)(row + kb);
  v8bf hi = *(const v8bf*)(row + kb + 16);
  return cat8(lo, hi);
}
__device__ __forceinline__ v16bf a_frag_f32(const float* row, int lane, int k0) {
  const int kb = k0 + ((lane < 16) ? 0 : 8);
  const float4 x0 = *(const float4*)(row + kb);
  const float4 x1 = *(const float4*)(row + kb + 4);
  const float4 y0 = *(const float4*)(row + kb + 16);
  const float4 y1 = *(const float4*)(row + kb + 20);
  v16bf a;
  a[0] = (__bf16)x0.x; a[1] = (__bf16)x0.y; a[2] = (__bf16)x0.z; a[3] = (__bf16)x0.w;
  a[4] = (__bf16)x1.x; a[5] = (__bf16)x1.y; a[6] = (__bf16)x1.z; a[7] = (__bf16)x1.w;
  a[8] = (__bf16)y0.x; a[9] = (__bf16)y0.y; a[10] = (__bf16)y0.z; a[11] = (__bf16)y0.w;
  a[12] = (__bf16)y1.x; a[13] = (__bf16)y1.y; a[14] = (__bf16)y1.z; a[15] = (__bf16)y1.w;
  return a;
}
__device__ __forceinline__ int a_kidx(int lane, int e) {
  return ((e < 8) ? e : (e + 8)) + ((lane < 16) ? 0 : 8);
}

// Stage 16 weight rows (fp32 -> bf16) into padded LDS slab, K elements each.
template <int K>
__device__ __forceinline__ void stage_wslab(const float* __restrict__ wbase, int row0,
                                            __bf16 (*slab)[K + 8], int tid) {
  const int col = tid & 15;
  const int kbase = (tid >> 4) * (K / 16);
  const float* wr = wbase + (size_t)(row0 + col) * K + kbase;
#pragma unroll
  for (int e = 0; e < K / 16; e += 4) {
    const float4 x = *(const float4*)(wr + e);
    slab[col][kbase + e + 0] = (__bf16)x.x;
    slab[col][kbase + e + 1] = (__bf16)x.y;
    slab[col][kbase + e + 2] = (__bf16)x.z;
    slab[col][kbase + e + 3] = (__bf16)x.w;
  }
}

// ======================= SE gate ============================================
__global__ __launch_bounds__(256)
void se_kernel(const float* __restrict__ query, const float* __restrict__ w1,
               const float* __restrict__ w2, float* __restrict__ se) {
  __shared__ float sm[CDIM];
  __shared__ float s1[CDIM / 2];
  const int b = blockIdx.x, c = threadIdx.x;
  float acc = 0.f;
  for (int n = 0; n < NTOK; ++n)
    acc += query[(size_t)n * (BD * CDIM) + b * CDIM + c];
  sm[c] = acc * (1.0f / NTOK);
  __syncthreads();
  if (c < 128) {
    float a = 0.f;
    for (int k = 0; k < CDIM; ++k) a += sm[k] * w1[c * CDIM + k];
    s1[c] = fmaxf(a, 0.f);
  }
  __syncthreads();
  float a = 0.f;
  for (int k = 0; k < 128; ++k) a += s1[k] * w2[c * 128 + k];
  se[b * CDIM + c] = 1.f / (1.f + expf(-a));
}

// ======================= conv 2x2/s2 as GEMM (M=B*Ns, K=4C, N=C) ============
__global__ __launch_bounds__(256)
void conv_wmma_kernel(const float* __restrict__ query, const float* __restrict__ sr_w,
                      const float* __restrict__ sr_b, float* __restrict__ xr_raw) {
  __shared__ __bf16 wslab[16][4 * CDIM + 8];
  const int lane = threadIdx.x & 31;
  const int tm = blockIdx.x * 8 + (threadIdx.x >> 5);
  const int tn = blockIdx.y;
  stage_wslab<4 * CDIM>(sr_w, tn * 16, wslab, threadIdx.x);
  __syncthreads();
  const int mrow = tm * 16 + (lane & 15);
  const int b = mrow >> 8;
  const int ns = mrow & 255;
  const int hs = ns >> 4, ws = ns & 15;
  const int ncol = tn * 16 + (lane & 15);
  v8f acc = {0.f, 0.f, 0.f, 0.f, 0.f, 0.f, 0.f, 0.f};
  for (int k0 = 0; k0 < 4 * CDIM; k0 += 32) {
    v16bf af, bf;
#pragma unroll
    for (int e = 0; e < 16; ++e) {
      const int k = k0 + a_kidx(lane, e);
      const int ci = k >> 2, i = (k >> 1) & 1, j = k & 1;
      const int tok = (2 * hs + i) * WW + (2 * ws + j);
      af[e] = (__bf16)query[(size_t)tok * (BD * CDIM) + b * CDIM + ci];
    }
    bf = b_frag_bf16(&wslab[ncol & 15][0], lane, k0);
    acc = __builtin_amdgcn_wmma_f32_16x16x32_bf16(false, af, false, bf,
                                                  (short)0, acc, false, false);
  }
#pragma unroll
  for (int r = 0; r < 8; ++r) {
    const int mo = tm * 16 + r + ((lane < 16) ? 0 : 8);
    const int no = tn * 16 + (lane & 15);
    xr_raw[(size_t)mo * CDIM + no] = acc[r] + sr_b[no];
  }
}

// ======================= per-row layernorm over C ===========================
__global__ __launch_bounds__(256)
void layernorm_kernel(const float* __restrict__ xin, const float* __restrict__ g,
                      const float* __restrict__ bb, float* __restrict__ xout) {
  __shared__ float red[CDIM];
  const int row = blockIdx.x, c = threadIdx.x;
  const float v = xin[(size_t)row * CDIM + c];
  red[c] = v;
  __syncthreads();
  for (int s = 128; s > 0; s >>= 1) { if (c < s) red[c] += red[c + s]; __syncthreads(); }
  const float mu = red[0] * (1.0f / CDIM);
  __syncthreads();
  const float d = v - mu;
  red[c] = d * d;
  __syncthreads();
  for (int s = 128; s > 0; s >>= 1) { if (c < s) red[c] += red[c + s]; __syncthreads(); }
  const float var = red[0] * (1.0f / CDIM);
  xout[(size_t)row * CDIM + c] = d * rsqrtf(var + 1e-5f) * g[c] + bb[c];
}

// ======================= q = relu((x Wq^T + bq)*scale), store (bh,n,d) bf16 =
__global__ __launch_bounds__(256)
void qproj_wmma_kernel(const float* __restrict__ query, const float* __restrict__ ipw,
                       const float* __restrict__ ipb, __bf16* __restrict__ q_bf) {
  __shared__ __bf16 wslab[16][CDIM + 8];
  const int lane = threadIdx.x & 31;
  const int tm = blockIdx.x * 8 + (threadIdx.x >> 5);
  const int tn = blockIdx.y;
  stage_wslab<CDIM>(ipw, tn * 16, wslab, threadIdx.x);
  __syncthreads();
  const int mrow = tm * 16 + (lane & 15);
  const int b = mrow >> 10, ntok = mrow & 1023;
  const int ncol = tn * 16 + (lane & 15);
  const float* Arow = query + (size_t)ntok * (BD * CDIM) + b * CDIM;
  v8f acc = {0.f, 0.f, 0.f, 0.f, 0.f, 0.f, 0.f, 0.f};
  for (int k0 = 0; k0 < CDIM; k0 += 32) {
    const v16bf af = a_frag_f32(Arow, lane, k0);
    const v16bf bf = b_frag_bf16(&wslab[ncol & 15][0], lane, k0);
    acc = __builtin_amdgcn_wmma_f32_16x16x32_bf16(false, af, false, bf,
                                                  (short)0, acc, false, false);
  }
  const float scaling = 0.17677669529663689f;  // hd^-0.5
#pragma unroll
  for (int r = 0; r < 8; ++r) {
    const int mo = tm * 16 + r + ((lane < 16) ? 0 : 8);
    const int bo = mo >> 10, nt = mo & 1023;
    const int no = tn * 16 + (lane & 15);
    const int h = no >> 5, d = no & 31;
    const float val = fmaxf((acc[r] + ipb[no]) * scaling, 0.f);
    q_bf[((size_t)(bo * HEADS + h) * NTOK + nt) * HD + d] = (__bf16)val;
  }
}

// ======= k = relu(xr Wk^T + bk); store reweighted TRANSPOSED (bh,128,Ns) ====
__global__ __launch_bounds__(256)
void kproj_wmma_kernel(const float* __restrict__ xr, const float* __restrict__ ipw,
                       const float* __restrict__ ipb, __bf16* __restrict__ kT_bf) {
  __shared__ __bf16 wslab[16][CDIM + 8];
  const int lane = threadIdx.x & 31;
  const int tm = blockIdx.x * 8 + (threadIdx.x >> 5);
  const int tn = blockIdx.y;
  stage_wslab<CDIM>(ipw, CDIM + tn * 16, wslab, threadIdx.x);
  __syncthreads();
  const int mrow = tm * 16 + (lane & 15);
  const int ncol = tn * 16 + (lane & 15);
  const float* Arow = xr + (size_t)mrow * CDIM;
  v8f acc = {0.f, 0.f, 0.f, 0.f, 0.f, 0.f, 0.f, 0.f};
  for (int k0 = 0; k0 < CDIM; k0 += 32) {
    const v16bf af = a_frag_f32(Arow, lane, k0);
    const v16bf bf = b_frag_bf16(&wslab[ncol & 15][0], lane, k0);
    acc = __builtin_amdgcn_wmma_f32_16x16x32_bf16(false, af, false, bf,
                                                  (short)0, acc, false, false);
  }
#pragma unroll
  for (int r = 0; r < 8; ++r) {
    const int mo = tm * 16 + r + ((lane < 16) ? 0 : 8);
    const int b = mo >> 8, ns = mo & 255;
    const int hs = ns >> 4, ws = ns & 15;
    const int no = tn * 16 + (lane & 15);
    const int h = no >> 5, d = no & 31;
    const float val = fmaxf(acc[r] + ipb[CDIM + no], 0.f);
    const float aa = PI2 * (float)hs * (1.0f / 16.f);
    const float bbb = PI2 * (float)ws * (1.0f / 16.f);
    const size_t base = (size_t)(b * HEADS + h) * DP * NS;
    kT_bf[base + (size_t)(0 * HD + d) * NS + ns] = (__bf16)(val * cosf(aa));
    kT_bf[base + (size_t)(1 * HD + d) * NS + ns] = (__bf16)(val * sinf(aa));
    kT_bf[base + (size_t)(2 * HD + d) * NS + ns] = (__bf16)(val * cosf(bbb));
    kT_bf[base + (size_t)(3 * HD + d) * NS + ns] = (__bf16)(val * sinf(bbb));
  }
}

// ========= v = xr Wv^T + bv; store TRANSPOSED (bh, d, ns) bf16 ==============
__global__ __launch_bounds__(256)
void vproj_wmma_kernel(const float* __restrict__ xr, const float* __restrict__ ipw,
                       const float* __restrict__ ipb, __bf16* __restrict__ vT_bf) {
  __shared__ __bf16 wslab[16][CDIM + 8];
  const int lane = threadIdx.x & 31;
  const int tm = blockIdx.x * 8 + (threadIdx.x >> 5);
  const int tn = blockIdx.y;
  stage_wslab<CDIM>(ipw, 2 * CDIM + tn * 16, wslab, threadIdx.x);
  __syncthreads();
  const int mrow = tm * 16 + (lane & 15);
  const int ncol = tn * 16 + (lane & 15);
  const float* Arow = xr + (size_t)mrow * CDIM;
  v8f acc = {0.f, 0.f, 0.f, 0.f, 0.f, 0.f, 0.f, 0.f};
  for (int k0 = 0; k0 < CDIM; k0 += 32) {
    const v16bf af = a_frag_f32(Arow, lane, k0);
    const v16bf bf = b_frag_bf16(&wslab[ncol & 15][0], lane, k0);
    acc = __builtin_amdgcn_wmma_f32_16x16x32_bf16(false, af, false, bf,
                                                  (short)0, acc, false, false);
  }
#pragma unroll
  for (int r = 0; r < 8; ++r) {
    const int mo = tm * 16 + r + ((lane < 16) ? 0 : 8);
    const int b = mo >> 8, ns = mo & 255;
    const int no = tn * 16 + (lane & 15);
    const int h = no >> 5, d = no & 31;
    vT_bf[((size_t)(b * HEADS + h) * HD + d) * NS + ns] =
        (__bf16)(acc[r] + ipb[2 * CDIM + no]);
  }
}

// ======================= ksum[bh][dp] = sum_ns kT[bh][dp][ns] ===============
__global__ __launch_bounds__(256)
void ksum_kernel(const __bf16* __restrict__ kT_bf, float* __restrict__ ksum) {
  const int idx = blockIdx.x * 256 + threadIdx.x;  // BH*DP = 65536
  const __bf16* p = kT_bf + (size_t)idx * NS;
  float s = 0.f;
  for (int n = 0; n < NS; n += 8) {
    const v8bf x = *(const v8bf*)(p + n);
#pragma unroll
    for (int e = 0; e < 8; ++e) s += (float)x[e];
  }
  ksum[idx] = s;
}

// ==== kv[bh] = kT(128xNs) @ v(Nsx32); store kvT (bh, 32, 128) bf16 ==========
__global__ __launch_bounds__(256)
void kv_wmma_kernel(const __bf16* __restrict__ kT_bf, const __bf16* __restrict__ vT_bf,
                    __bf16* __restrict__ kvT_bf) {
  const int lane = threadIdx.x & 31;
  const int tm = threadIdx.x >> 5;  // 8 M-tiles (DP=128)
  const int tn = blockIdx.x;        // 2
  const int bh = blockIdx.y;        // 512
  const __bf16* Arow = kT_bf + (size_t)bh * DP * NS + (size_t)(tm * 16 + (lane & 15)) * NS;
  const int ncol = tn * 16 + (lane & 15);
  const __bf16* Brow = vT_bf + ((size_t)bh * HD + ncol) * NS;  // K-contiguous
  v8f acc = {0.f, 0.f, 0.f, 0.f, 0.f, 0.f, 0.f, 0.f};
  for (int k0 = 0; k0 < NS; k0 += 32) {
    const v16bf af = a_frag_bf16(Arow, lane, k0);
    const v16bf bf = b_frag_bf16(Brow, lane, k0);
    acc = __builtin_amdgcn_wmma_f32_16x16x32_bf16(false, af, false, bf,
                                                  (short)0, acc, false, false);
  }
#pragma unroll
  for (int r = 0; r < 8; ++r) {
    const int mo = tm * 16 + r + ((lane < 16) ? 0 : 8);
    const int no = tn * 16 + (lane & 15);
    kvT_bf[((size_t)bh * HD + no) * DP + mo] = (__bf16)acc[r];
  }
}

// ============ denom[bh][n] = absclamp( sum_seg f_seg(n) * (q . ksum_seg) ) ==
__global__ __launch_bounds__(256)
void denom_kernel(const __bf16* __restrict__ q_bf, const float* __restrict__ ksum,
                  float* __restrict__ denom) {
  const int idx = blockIdx.x * 256 + threadIdx.x;  // BH*NTOK
  const int bh = idx >> 10, n = idx & 1023;
  const __bf16* q = q_bf + ((size_t)bh * NTOK + n) * HD;
  const float* ks = ksum + bh * DP;
  float qf[HD];
#pragma unroll
  for (int d0 = 0; d0 < HD; d0 += 8) {
    const v8bf x = *(const v8bf*)(q + d0);
#pragma unroll
    for (int e = 0; e < 8; ++e) qf[d0 + e] = (float)x[e];
  }
  const float aa = PI2 * (float)(n >> 5) * (1.0f / 32.f);
  const float bbb = PI2 * (float)(n & 31) * (1.0f / 32.f);
  const float f[4] = {cosf(aa), sinf(aa), cosf(bbb), sinf(bbb)};
  float s = 0.f;
#pragma unroll
  for (int seg = 0; seg < 4; ++seg) {
    float t = 0.f;
    for (int d = 0; d < HD; ++d) t += qf[d] * ks[seg * HD + d];
    s += f[seg] * t;
  }
  const float sgn = (s > 0.f) ? 1.f : ((s < 0.f) ? -1.f : 0.f);
  denom[idx] = fminf(fmaxf(fabsf(s), 1e-4f), 1e4f) * sgn;
}

// ===== attn[bh] = (sum_seg diag(f_seg) q) @ kv / denom ; write (B,N,C) bf16 =
__global__ __launch_bounds__(256)
void attn_wmma_kernel(const __bf16* __restrict__ q_bf, const __bf16* __restrict__ kvT_bf,
                      const float* __restrict__ denom, __bf16* __restrict__ attn_bf) {
  __shared__ __bf16 bslab[16][DP + 8];
  const int lane = threadIdx.x & 31;
  const int tm = blockIdx.x * 8 + (threadIdx.x >> 5);  // 64 M-tiles (NTOK)
  const int tn = blockIdx.y;                            // 2
  const int bh = blockIdx.z;                            // 512
  // --- stage 16x128 bf16 B slab (kvT rows tn*16..+15) into LDS via async DMA
  {
    const int t = threadIdx.x;
    const int r = t >> 4, c = t & 15;  // 16 rows x 16 chunks of 16B
    const __bf16* gsrc = kvT_bf + ((size_t)bh * HD + tn * 16 + r) * DP + c * 8;
    // Generic LDS addresses carry the LDS byte offset in addr[31:0] (ISA 10.2).
    const unsigned ldsa = (unsigned)(size_t)(const void*)&bslab[r][c * 8];
    asm volatile("global_load_async_to_lds_b128 %0, %1, off"
                 :
                 : "v"(ldsa), "v"(gsrc)
                 : "memory");
    asm volatile("s_wait_asynccnt 0x0" ::: "memory");
  }
  __syncthreads();
  const int mrow = tm * 16 + (lane & 15);
  const float aa = PI2 * (float)(mrow >> 5) * (1.0f / 32.f);
  const float bbb = PI2 * (float)(mrow & 31) * (1.0f / 32.f);
  const float f[4] = {cosf(aa), sinf(aa), cosf(bbb), sinf(bbb)};
  // q-row fragment slice is K-loop invariant; only the seg factor changes.
  const __bf16* qrow = q_bf + ((size_t)bh * NTOK + mrow) * HD;
  const int kb = (lane < 16) ? 0 : 8;
  const v8bf qlo = *(const v8bf*)(qrow + kb);
  const v8bf qhi = *(const v8bf*)(qrow + kb + 16);
  float qf[16];
#pragma unroll
  for (int e = 0; e < 8; ++e) { qf[e] = (float)qlo[e]; qf[e + 8] = (float)qhi[e]; }
  const int ncol = tn * 16 + (lane & 15);
  v8f acc = {0.f, 0.f, 0.f, 0.f, 0.f, 0.f, 0.f, 0.f};
#pragma unroll
  for (int k0 = 0; k0 < DP; k0 += 32) {
    const float fct = f[k0 >> 5];
    v16bf af;
#pragma unroll
    for (int e = 0; e < 16; ++e) af[e] = (__bf16)(qf[e] * fct);
    const v16bf bf = b_frag_bf16(&bslab[ncol & 15][0], lane, k0);
    acc = __builtin_amdgcn_wmma_f32_16x16x32_bf16(false, af, false, bf,
                                                  (short)0, acc, false, false);
  }
  const int b = bh >> 3, h = bh & 7;
#pragma unroll
  for (int r = 0; r < 8; ++r) {
    const int mo = tm * 16 + r + ((lane < 16) ? 0 : 8);
    const int no = tn * 16 + (lane & 15);
    const float val = acc[r] / denom[bh * NTOK + mo];
    attn_bf[((size_t)b * NTOK + mo) * CDIM + h * HD + no] = (__bf16)val;
  }
}

// ===== out = (attn outW^T + b) * se ; write transposed (N,B,C) fp32 =========
__global__ __launch_bounds__(256)
void outproj_wmma_kernel(const __bf16* __restrict__ attn_bf, const float* __restrict__ out_w,
                         const float* __restrict__ out_b, const float* __restrict__ se,
                         float* __restrict__ out) {
  __shared__ __bf16 wslab[16][CDIM + 8];
  const int lane = threadIdx.x & 31;
  const int tm = blockIdx.x * 8 + (threadIdx.x >> 5);
  const int tn = blockIdx.y;
  stage_wslab<CDIM>(out_w, tn * 16, wslab, threadIdx.x);
  __syncthreads();
  const int mrow = tm * 16 + (lane & 15);
  const int ncol = tn * 16 + (lane & 15);
  const __bf16* Arow = attn_bf + (size_t)mrow * CDIM;
  v8f acc = {0.f, 0.f, 0.f, 0.f, 0.f, 0.f, 0.f, 0.f};
  for (int k0 = 0; k0 < CDIM; k0 += 32) {
    const v16bf af = a_frag_bf16(Arow, lane, k0);
    const v16bf bf = b_frag_bf16(&wslab[ncol & 15][0], lane, k0);
    acc = __builtin_amdgcn_wmma_f32_16x16x32_bf16(false, af, false, bf,
                                                  (short)0, acc, false, false);
  }
#pragma unroll
  for (int r = 0; r < 8; ++r) {
    const int mo = tm * 16 + r + ((lane < 16) ? 0 : 8);
    const int b = mo >> 10, nt = mo & 1023;
    const int no = tn * 16 + (lane & 15);
    const float val = (acc[r] + out_b[no]) * se[b * CDIM + no];
    out[((size_t)nt * BD + b) * CDIM + no] = val;
  }
}

extern "C" void kernel_launch(void* const* d_in, const int* in_sizes, int n_in,
                              void* d_out, int out_size, void* d_ws, size_t ws_size,
                              hipStream_t stream) {
  const float* query = (const float*)d_in[0];
  const float* ipw  = (const float*)d_in[5];
  const float* ipb  = (const float*)d_in[6];
  const float* sr_w = (const float*)d_in[7];
  const float* sr_b = (const float*)d_in[8];
  const float* ng   = (const float*)d_in[9];
  const float* nb   = (const float*)d_in[10];
  const float* ow   = (const float*)d_in[11];
  const float* ob   = (const float*)d_in[12];
  const float* sw1  = (const float*)d_in[13];
  const float* sw2  = (const float*)d_in[14];
  float* out = (float*)d_out;

  char* w = (char*)d_ws;
  float*  xr_raw = (float*)w;  w += (size_t)BD * NS * CDIM * 4;
  float*  xr_ln  = (float*)w;  w += (size_t)BD * NS * CDIM * 4;
  float*  se     = (float*)w;  w += (size_t)BD * CDIM * 4;
  __bf16* q_bf   = (__bf16*)w; w += (size_t)BH * NTOK * HD * 2;
  __bf16* kT_bf  = (__bf16*)w; w += (size_t)BH * DP * NS * 2;
  __bf16* vT_bf  = (__bf16*)w; w += (size_t)BH * HD * NS * 2;
  __bf16* kvT_bf = (__bf16*)w; w += (size_t)BH * HD * DP * 2;
  float*  ksum   = (float*)w;  w += (size_t)BH * DP * 4;
  float*  denom  = (float*)w;  w += (size_t)BH * NTOK * 4;
  __bf16* attn_bf= (__bf16*)w; w += (size_t)BD * NTOK * CDIM * 2;

  se_kernel<<<BD, 256, 0, stream>>>(query, sw1, sw2, se);
  conv_wmma_kernel<<<dim3(128, 16), 256, 0, stream>>>(query, sr_w, sr_b, xr_raw);
  layernorm_kernel<<<BD * NS, 256, 0, stream>>>(xr_raw, ng, nb, xr_ln);
  qproj_wmma_kernel<<<dim3(512, 16), 256, 0, stream>>>(query, ipw, ipb, q_bf);
  kproj_wmma_kernel<<<dim3(128, 16), 256, 0, stream>>>(xr_ln, ipw, ipb, kT_bf);
  vproj_wmma_kernel<<<dim3(128, 16), 256, 0, stream>>>(xr_ln, ipw, ipb, vT_bf);
  ksum_kernel<<<(BH * DP) / 256, 256, 0, stream>>>(kT_bf, ksum);
  kv_wmma_kernel<<<dim3(2, BH), 256, 0, stream>>>(kT_bf, vT_bf, kvT_bf);
  denom_kernel<<<(BH * NTOK) / 256, 256, 0, stream>>>(q_bf, ksum, denom);
  attn_wmma_kernel<<<dim3(8, 2, BH), 256, 0, stream>>>(q_bf, kvT_bf, denom, attn_bf);
  outproj_wmma_kernel<<<dim3(512, 16), 256, 0, stream>>>(attn_bf, ow, ob, se, out);
}
